// SuperGAT_18021682774393
// MI455X (gfx1250) — matrix operand order repaired and checked by reading, not compiled, not run on hardware
//
#include <hip/hip_runtime.h>

// ---------------------------------------------------------------------------
// Types for CDNA5 WMMA (wave32)
// ---------------------------------------------------------------------------
typedef __attribute__((ext_vector_type(16))) __bf16 v16bf;
typedef __attribute__((ext_vector_type(8)))  __bf16 v8bf;
typedef __attribute__((ext_vector_type(8)))  float  v8f;

#define H_DIM 128
#define IN_DIM 256
#define BN_EPS 1e-5f
#define SLOPE 0.2f

// ---------------------------------------------------------------------------
// Helpers
// ---------------------------------------------------------------------------
static __device__ __forceinline__ unsigned short f2bf(float f) {
    unsigned u = __float_as_uint(f);
    u += 0x7fffu + ((u >> 16) & 1u);           // round-to-nearest-even
    return (unsigned short)(u >> 16);
}

// order-preserving float <-> uint key for atomic segment-max
static __device__ __forceinline__ unsigned fkey(float f) {
    unsigned u = __float_as_uint(f);
    return (u & 0x80000000u) ? ~u : (u | 0x80000000u);
}
static __device__ __forceinline__ float keyf(unsigned k) {
    unsigned u = (k & 0x80000000u) ? (k & 0x7fffffffu) : ~k;
    return __uint_as_float(u);
}

// ---------------------------------------------------------------------------
// Prep kernels: f32 -> bf16, and weight transpose W[K][128] -> Wt[128][K]
// ---------------------------------------------------------------------------
__global__ void f32_to_bf16_kernel(const float* __restrict__ src,
                                   unsigned short* __restrict__ dst, long n) {
    long i = (long)blockIdx.x * blockDim.x + threadIdx.x;
    if (i < n) dst[i] = f2bf(src[i]);
}

__global__ void transpose_bf16_kernel(const float* __restrict__ W,
                                      unsigned short* __restrict__ Wt, int K) {
    int i = blockIdx.x * blockDim.x + threadIdx.x;  // over K*128
    if (i >= K * H_DIM) return;
    int k = i / H_DIM, c = i % H_DIM;
    Wt[c * K + k] = f2bf(W[k * H_DIM + c]);
}

// ---------------------------------------------------------------------------
// WMMA GEMM: C[n][128] = A_bf16[n][K] @ Bt_bf16[128][K]
// One block = 16 rows; 8 waves, each wave owns a 16-column tile.
// mode 0: store raw f32 to Cf32.
// mode 1: fused  relu(bn(val + bias))  -> Cf32 and bf16 Cbf.
// ---------------------------------------------------------------------------
__global__ void __launch_bounds__(256)
gemm_bf16_wmma(const unsigned short* __restrict__ A,
               const unsigned short* __restrict__ Bt,
               int n, int K,
               const float* __restrict__ bias, const float* __restrict__ g,
               const float* __restrict__ be,   const float* __restrict__ m,
               const float* __restrict__ v,    int mode,
               float* __restrict__ Cf32, unsigned short* __restrict__ Cbf) {
    const int tile = blockIdx.x;
    const int wave = threadIdx.x >> 5;
    const int lane = threadIdx.x & 31;
    const int r = lane & 15;       // local row for A operand
    const int q = lane >> 4;       // half-wave select
    const int gc = wave * 16 + r;  // global output column (0..127)

    int gr = tile * 16 + r;
    int ar = (gr < n) ? gr : 0;    // clamp (uniform-safe, no divergence)
    const unsigned short* Arow = A + (size_t)ar * K;
    const unsigned short* Brow = Bt + (size_t)gc * K;

    v8f acc = {};
    for (int k0 = 0; k0 < K; k0 += 32) {
        // A 16x32 bf16: half-wave q holds K = q*16..q*16+15
        v8bf alo = *(const v8bf*)(Arow + k0 + q * 16);
        v8bf ahi = *(const v8bf*)(Arow + k0 + q * 16 + 8);
        v16bf a;
#pragma unroll
        for (int t = 0; t < 8; ++t) { a[t] = alo[t]; a[t + 8] = ahi[t]; }
        // B 32x16 bf16: lane = column, half-wave q holds K = q*16..q*16+15
        v16bf b = *(const v16bf*)(Brow + k0 + q * 16);
        acc = __builtin_amdgcn_wmma_f32_16x16x32_bf16(
            false, a, false, b, (short)0, acc, false, false);
    }

    float bi = 0.f, scale = 1.f, shift = 0.f;
    if (mode == 1) {
        bi = bias[gc];
        scale = g[gc] * rsqrtf(v[gc] + BN_EPS);
        shift = be[gc] - m[gc] * scale;
    }
#pragma unroll
    for (int j = 0; j < 8; ++j) {
        int row = tile * 16 + j + 8 * q;     // C/D layout: VGPR j -> M = j + 8*q
        if (row < n) {
            float val = acc[j];
            if (mode == 1) {
                val = (val + bi) * scale + shift;
                val = fmaxf(val, 0.f);
                Cbf[(size_t)row * H_DIM + gc] = f2bf(val);
            }
            Cf32[(size_t)row * H_DIM + gc] = val;
        }
    }
}

// ---------------------------------------------------------------------------
// Per-layer init: agg = 0, amax key = 0 (== -inf under key order), den = 0
// ---------------------------------------------------------------------------
__global__ void init_kernel(float* __restrict__ agg, unsigned* __restrict__ amax,
                            float* __restrict__ den, int n) {
    long i = (long)blockIdx.x * blockDim.x + threadIdx.x;
    if (i < (long)n * H_DIM) agg[i] = 0.f;
    if (i < n) { amax[i] = 0u; den[i] = 0.f; }
}

// ---------------------------------------------------------------------------
// Edge pass A: one wave per edge. alpha = leakyrelu((hj.attl + hi.attr) *
// sigmoid(hi.hj)); segment-max via ordered-uint atomicMax.
// Self-loops are the last n edges (e >= E -> node e-E).
// ---------------------------------------------------------------------------
__global__ void __launch_bounds__(256)
edge_pass_a(const float* __restrict__ h, const int* __restrict__ ei,
            int E, int n, const float* __restrict__ attl,
            const float* __restrict__ attr, float* __restrict__ alphaBuf,
            unsigned* __restrict__ amax) {
    const int Et = E + n;
    int e = blockIdx.x * 8 + (threadIdx.x >> 5);
    if (e >= Et) return;
    const int lane = threadIdx.x & 31;
    int s = (e < E) ? ei[e] : (e - E);
    int d = (e < E) ? ei[E + e] : (e - E);

    float4 hj = *(const float4*)(h + (size_t)s * H_DIM + lane * 4);
    float4 hi = *(const float4*)(h + (size_t)d * H_DIM + lane * 4);
    float4 al = *(const float4*)(attl + lane * 4);
    float4 ar = *(const float4*)(attr + lane * 4);

    float lg = hj.x * hi.x + hj.y * hi.y + hj.z * hi.z + hj.w * hi.w;
    float as = hj.x * al.x + hj.y * al.y + hj.z * al.z + hj.w * al.w
             + hi.x * ar.x + hi.y * ar.y + hi.z * ar.z + hi.w * ar.w;
#pragma unroll
    for (int o = 16; o; o >>= 1) {
        lg += __shfl_xor(lg, o, 32);
        as += __shfl_xor(as, o, 32);
    }
    if (lane == 0) {
        float sig = 1.f / (1.f + expf(-lg));
        float a = as * sig;
        a = (a > 0.f) ? a : SLOPE * a;
        alphaBuf[e] = a;
        atomicMax(amax + d, fkey(a));
    }
}

// ---------------------------------------------------------------------------
// Edge pass B: one thread per edge. ex = exp(alpha - amax[dst]); den += ex.
// alphaBuf is overwritten with ex in place.
// ---------------------------------------------------------------------------
__global__ void edge_pass_b(const int* __restrict__ ei, int E, int n,
                            float* __restrict__ alphaBuf,
                            const unsigned* __restrict__ amax,
                            float* __restrict__ den) {
    const int Et = E + n;
    int e = blockIdx.x * blockDim.x + threadIdx.x;
    if (e >= Et) return;
    int d = (e < E) ? ei[E + e] : (e - E);
    float ex = expf(alphaBuf[e] - keyf(amax[d]));
    alphaBuf[e] = ex;
    atomicAdd(den + d, ex);
}

// ---------------------------------------------------------------------------
// Edge pass C: one wave per edge. agg[dst] += hj * (ex/den[dst]).
// ---------------------------------------------------------------------------
__global__ void __launch_bounds__(256)
edge_pass_c(const float* __restrict__ h, const int* __restrict__ ei,
            int E, int n, const float* __restrict__ exBuf,
            const float* __restrict__ den, float* __restrict__ agg) {
    const int Et = E + n;
    int e = blockIdx.x * 8 + (threadIdx.x >> 5);
    if (e >= Et) return;
    const int lane = threadIdx.x & 31;
    int s = (e < E) ? ei[e] : (e - E);
    int d = (e < E) ? ei[E + e] : (e - E);
    float a = exBuf[e] / den[d];
    float4 hj = *(const float4*)(h + (size_t)s * H_DIM + lane * 4);
    float* o = agg + (size_t)d * H_DIM + lane * 4;
    atomicAdd(o + 0, hj.x * a);
    atomicAdd(o + 1, hj.y * a);
    atomicAdd(o + 2, hj.z * a);
    atomicAdd(o + 3, hj.w * a);
}

// ---------------------------------------------------------------------------
// Node epilogue: relu(bn(agg + bias)) -> bf16 (input to next GEMM)
// ---------------------------------------------------------------------------
__global__ void node_epilogue(const float* __restrict__ agg,
                              const float* __restrict__ bias,
                              const float* __restrict__ g,
                              const float* __restrict__ be,
                              const float* __restrict__ m,
                              const float* __restrict__ v, int n,
                              unsigned short* __restrict__ postb) {
    long i = (long)blockIdx.x * blockDim.x + threadIdx.x;
    if (i >= (long)n * H_DIM) return;
    int f = (int)(i & (H_DIM - 1));
    float val = agg[i] + bias[f];
    val = (val - m[f]) * rsqrtf(v[f] + BN_EPS) * g[f] + be[f];
    val = fmaxf(val, 0.f);
    postb[i] = f2bf(val);
}

// ---------------------------------------------------------------------------
// fc3: out[n][2] = h[n][128] @ fcw3[128][2] + fcb3
// ---------------------------------------------------------------------------
__global__ void fc3_kernel(const float* __restrict__ h,
                           const float* __restrict__ w,
                           const float* __restrict__ b,
                           float* __restrict__ out, int n) {
    int i = blockIdx.x * blockDim.x + threadIdx.x;
    if (i >= n) return;
    const float* hr = h + (size_t)i * H_DIM;
    float a0 = b[0], a1 = b[1];
#pragma unroll 4
    for (int k = 0; k < H_DIM; ++k) {
        float hv = hr[k];
        a0 += hv * w[2 * k];
        a1 += hv * w[2 * k + 1];
    }
    out[2 * i] = a0;
    out[2 * i + 1] = a1;
}

// ---------------------------------------------------------------------------
// Host launcher
// ---------------------------------------------------------------------------
extern "C" void kernel_launch(void* const* d_in, const int* in_sizes, int n_in,
                              void* d_out, int out_size, void* d_ws, size_t ws_size,
                              hipStream_t stream) {
    const float* x    = (const float*)d_in[0];
    const int*   ei   = (const int*)d_in[1];
    const float* W1   = (const float*)d_in[2];
    const float* b1   = (const float*)d_in[3];
    const float* attl1= (const float*)d_in[4];
    const float* attr1= (const float*)d_in[5];
    const float* g1   = (const float*)d_in[6];
    const float* be1  = (const float*)d_in[7];
    const float* m1   = (const float*)d_in[8];
    const float* v1   = (const float*)d_in[9];
    const float* W2   = (const float*)d_in[10];
    const float* b2   = (const float*)d_in[11];
    const float* attl2= (const float*)d_in[12];
    const float* attr2= (const float*)d_in[13];
    const float* g2   = (const float*)d_in[14];
    const float* be2  = (const float*)d_in[15];
    const float* m2   = (const float*)d_in[16];
    const float* v2   = (const float*)d_in[17];
    const float* fcw1 = (const float*)d_in[18];
    const float* fcb1 = (const float*)d_in[19];
    const float* g3   = (const float*)d_in[20];
    const float* be3  = (const float*)d_in[21];
    const float* m3   = (const float*)d_in[22];
    const float* v3   = (const float*)d_in[23];
    const float* fcw2 = (const float*)d_in[24];
    const float* fcb2 = (const float*)d_in[25];
    const float* g4   = (const float*)d_in[26];
    const float* be4  = (const float*)d_in[27];
    const float* m4   = (const float*)d_in[28];
    const float* v4   = (const float*)d_in[29];
    const float* fcw3 = (const float*)d_in[30];
    const float* fcb3 = (const float*)d_in[31];
    float* out = (float*)d_out;

    const int N  = in_sizes[0] / IN_DIM;
    const int E  = in_sizes[1] / 2;
    const int Et = E + N;

    // ---- workspace carve-up (256B aligned) ----
    char* wsp = (char*)d_ws;
    size_t off = 0;
    auto take = [&](size_t bytes) -> char* {
        char* p = wsp + off;
        off = (off + bytes + 255) & ~(size_t)255;
        return p;
    };
    unsigned short* xb    = (unsigned short*)take((size_t)N * IN_DIM * 2);
    unsigned short* w1t   = (unsigned short*)take((size_t)IN_DIM * H_DIM * 2);
    unsigned short* w2t   = (unsigned short*)take((size_t)H_DIM * H_DIM * 2);
    unsigned short* fc1t  = (unsigned short*)take((size_t)H_DIM * H_DIM * 2);
    unsigned short* fc2t  = (unsigned short*)take((size_t)H_DIM * H_DIM * 2);
    float*          hraw  = (float*)take((size_t)N * H_DIM * 4);
    float*          agg   = (float*)take((size_t)N * H_DIM * 4);
    float*          post2 = (float*)take((size_t)N * H_DIM * 4);
    unsigned short* postb = (unsigned short*)take((size_t)N * H_DIM * 2);
    unsigned short* postb2= (unsigned short*)take((size_t)N * H_DIM * 2);
    float*          alpha = (float*)take((size_t)Et * 4);
    unsigned*       amax  = (unsigned*)take((size_t)N * 4);
    float*          den   = (float*)take((size_t)N * 4);
    (void)ws_size; (void)n_in; (void)out_size;

    const int T = 256;
    dim3 gemmGrid((N + 15) / 16);
    int edgeWaveBlocks = (Et + 7) / 8;
    int edgeThrBlocks  = (Et + T - 1) / T;
    int nodeFeatBlocks = ((long)N * H_DIM + T - 1) / T;

    // ---- prep: bf16 conversions / transposes ----
    f32_to_bf16_kernel<<<((long)N * IN_DIM + T - 1) / T, T, 0, stream>>>(
        x, xb, (long)N * IN_DIM);
    transpose_bf16_kernel<<<(IN_DIM * H_DIM + T - 1) / T, T, 0, stream>>>(W1, w1t, IN_DIM);
    transpose_bf16_kernel<<<(H_DIM * H_DIM + T - 1) / T, T, 0, stream>>>(W2, w2t, H_DIM);
    transpose_bf16_kernel<<<(H_DIM * H_DIM + T - 1) / T, T, 0, stream>>>(fcw1, fc1t, H_DIM);
    transpose_bf16_kernel<<<(H_DIM * H_DIM + T - 1) / T, T, 0, stream>>>(fcw2, fc2t, H_DIM);

    // ---- layer 1: h = x @ W1, SuperGAT attention, BN+ReLU ----
    gemm_bf16_wmma<<<gemmGrid, T, 0, stream>>>(xb, w1t, N, IN_DIM,
        nullptr, nullptr, nullptr, nullptr, nullptr, 0, hraw, nullptr);
    init_kernel<<<nodeFeatBlocks, T, 0, stream>>>(agg, amax, den, N);
    edge_pass_a<<<edgeWaveBlocks, T, 0, stream>>>(hraw, ei, E, N, attl1, attr1, alpha, amax);
    edge_pass_b<<<edgeThrBlocks, T, 0, stream>>>(ei, E, N, alpha, amax, den);
    edge_pass_c<<<edgeWaveBlocks, T, 0, stream>>>(hraw, ei, E, N, alpha, den, agg);
    node_epilogue<<<nodeFeatBlocks, T, 0, stream>>>(agg, b1, g1, be1, m1, v1, N, postb);

    // ---- layer 2 ----
    gemm_bf16_wmma<<<gemmGrid, T, 0, stream>>>(postb, w2t, N, H_DIM,
        nullptr, nullptr, nullptr, nullptr, nullptr, 0, hraw, nullptr);
    init_kernel<<<nodeFeatBlocks, T, 0, stream>>>(agg, amax, den, N);
    edge_pass_a<<<edgeWaveBlocks, T, 0, stream>>>(hraw, ei, E, N, attl2, attr2, alpha, amax);
    edge_pass_b<<<edgeThrBlocks, T, 0, stream>>>(ei, E, N, alpha, amax, den);
    edge_pass_c<<<edgeWaveBlocks, T, 0, stream>>>(hraw, ei, E, N, alpha, den, agg);
    node_epilogue<<<nodeFeatBlocks, T, 0, stream>>>(agg, b2, g2, be2, m2, v2, N, postb);

    // ---- MLP head: fc1, fc2 fused BN+ReLU; fc3 plain ----
    gemm_bf16_wmma<<<gemmGrid, T, 0, stream>>>(postb, fc1t, N, H_DIM,
        fcb1, g3, be3, m3, v3, 1, post2, postb2);
    gemm_bf16_wmma<<<gemmGrid, T, 0, stream>>>(postb2, fc2t, N, H_DIM,
        fcb2, g4, be4, m4, v4, 1, post2, postb);
    fc3_kernel<<<(N + T - 1) / T, T, 0, stream>>>(post2, fcw3, fcb3, out, N);
}
